// DeepseekV2MoE_45019847197158
// MI455X (gfx1250) — matrix operand order repaired and checked by reading, not compiled, not run on hardware
//
#include <hip/hip_runtime.h>
#include <cstddef>

// ---------------------------------------------------------------------------
// DeepseekV2 MoE for MI455X (gfx1250, wave32, WMMA bf16 16x16x32,
// async global->LDS staging, fp32 HW atomics for the combine-scatter)
//   k_init   : zero per-expert token counters (in d_ws)
//   k_gate   : softmax routing, group-limited top-6, compacted expert lists
//   k_shared : fused shared-expert FFN, writes out (covers every element)
//   k_routed : per-(expert, token-tile) fused FFN, atomic-adds into out
// ws layout: [cnt: E ints][pad to 256B][list: E*T ints][wlist: E*T floats]
// ---------------------------------------------------------------------------

typedef __attribute__((ext_vector_type(16))) __bf16 v16bf;
typedef __attribute__((ext_vector_type(8)))  __bf16 v8bf;
typedef __attribute__((ext_vector_type(8)))  float  v8f;
typedef int v4i __attribute__((vector_size(16)));   // matches builtin's param type

#define H_DIM   2048
#define F_DIM   1024
#define FS_DIM  2048   // N_SHARED * F
#define E_NUM   32
#define G_NUM   8
#define TOPKG   3
#define TOP_K   6
#define T_NUM   2048
#define SCALE_F 1.0f

#define MT_R 64                 // routed token tile
#define MT_S 32                 // shared token tile
#define XPF  36                 // f32 X stage pitch (dwords): 16B rows, bank-spread
#define WPF  132                // f32 W stage pitch (dwords): 16B rows, bank-spread
#define AP   (F_DIM + 8)        // routed act pitch (bf16 elems)
#define APS  (FS_DIM + 8)       // shared act pitch (bf16 elems)

// ---- CDNA5 async global->LDS copy (ASYNCcnt) with sync fallback -----------
#if defined(__has_builtin)
#if __has_builtin(__builtin_amdgcn_global_load_async_to_lds_b128)
#define ASYNC_LDS 1
#endif
#endif

__device__ inline void cp16_g2l(const float* g, float* l) {
#ifdef ASYNC_LDS
  __builtin_amdgcn_global_load_async_to_lds_b128(
      (__attribute__((address_space(1))) v4i*)g,
      (__attribute__((address_space(3))) v4i*)l, 0, 0);
#else
  *(float4*)l = *(const float4*)g;
#endif
}

__device__ inline void async_wait0() {
#ifdef ASYNC_LDS
#if __has_builtin(__builtin_amdgcn_s_wait_asynccnt)
  __builtin_amdgcn_s_wait_asynccnt(0);
#else
  asm volatile("s_wait_asynccnt 0" ::: "memory");
#endif
#endif
}

__device__ inline float fast_rcp(float x) {
#if defined(__has_builtin) && __has_builtin(__builtin_amdgcn_rcpf)
  return __builtin_amdgcn_rcpf(x);   // v_rcp_f32
#else
  return 1.0f / x;
#endif
}

// ---- WMMA fragment helpers (wave32; contiguous k = 16*(lane/16) + s in A and B) ----------

// A 16x32 bf16 from bf16 LDS: lane<16 -> row=lane,    K = {0..15}
//                             lane>=16 -> row=lane-16, K = {16..31}
__device__ inline v16bf ldA_bf(const __bf16* Xs, int pitch, int rowbase, int kloc) {
  const int lane = threadIdx.x & 31;
  const int r  = rowbase + (lane & 15);
  const int k0 = kloc + ((lane >> 4) << 4);
  v8bf lo = *(const v8bf*)(Xs + r * pitch + k0);
  v8bf hi = *(const v8bf*)(Xs + r * pitch + k0 + 8);
  v16bf a;
#pragma unroll
  for (int i = 0; i < 8; ++i) { a[i] = lo[i]; a[i + 8] = hi[i]; }
  return a;
}

// A 16x32 bf16 built from f32 LDS tile (async-staged), cvt on read
__device__ inline v16bf ldA_f32(const float* Xs, int pitch, int rowbase) {
  const int lane = threadIdx.x & 31;
  const float* p = Xs + (size_t)(rowbase + (lane & 15)) * pitch + ((lane >> 4) << 4);
  float4 a0 = *(const float4*)(p);
  float4 a1 = *(const float4*)(p + 4);
  float4 b0 = *(const float4*)(p + 8);
  float4 b1 = *(const float4*)(p + 12);
  v16bf a;
  a[0] = (__bf16)a0.x; a[1] = (__bf16)a0.y; a[2]  = (__bf16)a0.z; a[3]  = (__bf16)a0.w;
  a[4] = (__bf16)a1.x; a[5] = (__bf16)a1.y; a[6]  = (__bf16)a1.z; a[7]  = (__bf16)a1.w;
  a[8] = (__bf16)b0.x; a[9] = (__bf16)b0.y; a[10] = (__bf16)b0.z; a[11] = (__bf16)b0.w;
  a[12]= (__bf16)b1.x; a[13]= (__bf16)b1.y; a[14] = (__bf16)b1.z; a[15] = (__bf16)b1.w;
  return a;
}

// B 32x16 bf16 from f32 LDS tile [32 x 128] (async-staged, pitch WPF)
__device__ inline v16bf ldB_lds(const float* Ws, int ncol_local) {
  const int lane = threadIdx.x & 31;
  const float* p = Ws + (size_t)((lane >> 4) << 4) * WPF + ncol_local + (lane & 15);
  v16bf b;
#pragma unroll
  for (int j = 0; j < 16; ++j) b[j] = (__bf16)p[j * WPF];
  return b;
}

// B 32x16 bf16, W stored [N,K] row-major (ld = ldk): contiguous per-lane gather
__device__ inline v16bf ldB_nk(const float* __restrict__ W, int ldk, int krow, int ncol) {
  const int lane = threadIdx.x & 31;
  const float* p = W + (size_t)(ncol + (lane & 15)) * ldk + krow + ((lane >> 4) << 4);
  v16bf b;
#pragma unroll
  for (int j = 0; j < 16; ++j) b[j] = (__bf16)p[j];
  return b;
}

__device__ inline v8f wmma_bf16(v16bf a, v16bf b, v8f c) {
  return __builtin_amdgcn_wmma_f32_16x16x32_bf16(false, a, false, b, (short)0, c,
                                                 false, false);
}

// ---------------------------------------------------------------------------
__global__ void k_init(int* cnt) {
  if (threadIdx.x < E_NUM) cnt[threadIdx.x] = 0;
}

// one wave32 per token; lane == expert
__global__ __launch_bounds__(256) void k_gate(const float* __restrict__ x,
                                              const float* __restrict__ wgate,
                                              int* __restrict__ cnt,
                                              int* __restrict__ list,
                                              float* __restrict__ wlist) {
  const int lane = threadIdx.x & 31;
  const int wid  = threadIdx.x >> 5;
  const int t    = blockIdx.x * 8 + wid;

  const float4* xv = (const float4*)(x + (size_t)t * H_DIM);
  const float4* wv = (const float4*)(wgate + (size_t)lane * H_DIM);
  float acc = 0.f;
  for (int i = 0; i < H_DIM / 4; ++i) {
    float4 a = xv[i], b = wv[i];
    acc += a.x * b.x + a.y * b.y + a.z * b.z + a.w * b.w;
  }
  // softmax across the 32 experts (one per lane)
  float m = acc;
  for (int off = 16; off; off >>= 1) m = fmaxf(m, __shfl_xor(m, off, 32));
  float ex = __expf(acc - m);
  float s = ex;
  for (int off = 16; off; off >>= 1) s += __shfl_xor(s, off, 32);
  float p = ex * fast_rcp(s);

  // group max over aligned groups of 4 lanes (E/G = 4)
  float gm = fmaxf(p, __shfl_xor(p, 1, 32));
  gm = fmaxf(gm, __shfl_xor(gm, 2, 32));
  const int myg = lane >> 2;
  int grank = 0;
#pragma unroll
  for (int g = 0; g < G_NUM; ++g) {
    float og = __shfl(gm, g * 4, 32);
    grank += (og > gm) || (og == gm && g < myg);
  }
  float v = (grank < TOPKG) ? p : 0.f;

  // top-6 of 32 by rank
  int rank = 0;
  for (int j = 0; j < 32; ++j) {
    float ov = __shfl(v, j, 32);
    rank += (ov > v) || (ov == v && j < lane);
  }
  const bool sel = (rank < TOP_K) && (v > 0.f);
  float sv = sel ? v : 0.f;
  float ssum = sv;
  for (int off = 16; off; off >>= 1) ssum += __shfl_xor(ssum, off, 32);
  const float w = sel ? v * fast_rcp(ssum + 1e-20f) * SCALE_F : 0.f;

  if (sel) {
    int pos = atomicAdd(&cnt[lane], 1);
    list[lane * T_NUM + pos]  = t;
    wlist[lane * T_NUM + pos] = w;
  }
}

// ---------------------------------------------------------------------------
// shared experts: out[t0..t0+31, :] = (silu(x sg^T) * (x su^T)) sd^T
__global__ __launch_bounds__(256) void k_shared(const float* __restrict__ x,
                                                const float* __restrict__ sg,
                                                const float* __restrict__ su,
                                                const float* __restrict__ sd,
                                                float* __restrict__ out) {
  __shared__ float  Xs[MT_S * XPF];
  __shared__ __bf16 Act[MT_S * APS];
  const int tid = threadIdx.x, lane = tid & 31, wid = tid >> 5;
  const int t0 = blockIdx.x * MT_S;

  // ---- phase 1: act = silu(x sg^T) * (x su^T)  (bf16 into LDS) ----
  for (int nb = 0; nb < FS_DIM; nb += 128) {
    const int n16 = nb + wid * 16;
    v8f accG[2] = {}, accU[2] = {};
    for (int kb = 0; kb < H_DIM; kb += 32) {
      __syncthreads();
      {                                   // 256 chunks of 16B, 1 per thread
        const int row = tid >> 3, c = (tid & 7) << 2;
        cp16_g2l(x + (size_t)(t0 + row) * H_DIM + kb + c, Xs + row * XPF + c);
      }
      async_wait0();
      __syncthreads();
      __builtin_prefetch(sg + (size_t)n16 * H_DIM + kb + 64, 0, 1);
      __builtin_prefetch(su + (size_t)n16 * H_DIM + kb + 64, 0, 1);
      v16bf bg = ldB_nk(sg, H_DIM, kb, n16);
      v16bf bu = ldB_nk(su, H_DIM, kb, n16);
#pragma unroll
      for (int ms = 0; ms < 2; ++ms) {
        v16bf a = ldA_f32(Xs, XPF, ms * 16);
        accG[ms] = wmma_bf16(a, bg, accG[ms]);
        accU[ms] = wmma_bf16(a, bu, accU[ms]);
      }
    }
#pragma unroll
    for (int ms = 0; ms < 2; ++ms)
#pragma unroll
      for (int r = 0; r < 8; ++r) {
        const int lm = ms * 16 + ((lane >> 4) << 3) + r;
        float g = accG[ms][r], u = accU[ms][r];
        float a = g * fast_rcp(1.f + __expf(-g)) * u;    // silu via v_rcp_f32
        Act[lm * APS + n16 + (lane & 15)] = (__bf16)a;
      }
  }
  __syncthreads();

  // ---- phase 2: out = act sd^T  (plain stores; block owns its rows) ----
  for (int hb = 0; hb < H_DIM; hb += 128) {
    const int n16 = hb + wid * 16;
    v8f acc[2] = {};
    for (int kb = 0; kb < FS_DIM; kb += 32) {
      __builtin_prefetch(sd + (size_t)n16 * FS_DIM + kb + 64, 0, 1);
      v16bf b = ldB_nk(sd, FS_DIM, kb, n16);   // sd is [H, 2F]: B[f,h]=sd[h*2F+f]
#pragma unroll
      for (int ms = 0; ms < 2; ++ms) {
        v16bf a = ldA_bf(Act, APS, ms * 16, kb);
        acc[ms] = wmma_bf16(a, b, acc[ms]);
      }
    }
#pragma unroll
    for (int ms = 0; ms < 2; ++ms)
#pragma unroll
      for (int r = 0; r < 8; ++r) {
        const int lm = ms * 16 + ((lane >> 4) << 3) + r;
        out[(size_t)(t0 + lm) * H_DIM + n16 + (lane & 15)] = acc[ms][r];
      }
  }
}

// ---------------------------------------------------------------------------
// routed experts: per (expert, 64-token tile) fused gate/up/silu/down,
// combine weight folded before down-proj, atomic-add scatter into out.
// X tile and W tiles async-staged into LDS each k-step.
__global__ __launch_bounds__(256) void k_routed(const float* __restrict__ x,
                                                const float* __restrict__ wg,
                                                const float* __restrict__ wu,
                                                const float* __restrict__ wd,
                                                const int* __restrict__ cnt,
                                                const int* __restrict__ list,
                                                const float* __restrict__ wlist,
                                                float* __restrict__ out) {
  __shared__ float  Xs[MT_R * XPF];          //  9.2 KB
  __shared__ float  Wst[2 * 32 * WPF];       // 33.8 KB (Wg|Wu, or Wd in half 0)
  __shared__ __bf16 Act[MT_R * AP];          // 129 KB
  __shared__ int    Tid[MT_R];
  __shared__ float  Twt[MT_R];

  const int e  = blockIdx.x;
  const int mt = blockIdx.y;
  const int n  = cnt[e];
  if (mt * MT_R >= n) return;                    // block-uniform early exit
  const int tid = threadIdx.x, lane = tid & 31, wid = tid >> 5;

  if (tid < MT_R) {
    const int idx = mt * MT_R + tid;
    if (idx < n) { Tid[tid] = list[e * T_NUM + idx]; Twt[tid] = wlist[e * T_NUM + idx]; }
    else         { Tid[tid] = 0;                     Twt[tid] = 0.f; }
  }
  __syncthreads();

  const float* Wg = wg + (size_t)e * H_DIM * F_DIM;
  const float* Wu = wu + (size_t)e * H_DIM * F_DIM;
  const float* Wd = wd + (size_t)e * F_DIM * H_DIM;
  float* Wgs = Wst;
  float* Wus = Wst + 32 * WPF;

  // ---- phase 1: act = silu(x Wg) * (x Wu) * combine_w ----
  for (int nb = 0; nb < F_DIM; nb += 128) {
    v8f accG[4] = {}, accU[4] = {};
    for (int kb = 0; kb < H_DIM; kb += 32) {
      __syncthreads();
      // X tile: 64x32 f32 = 512 16B-chunks (gathered rows)
      for (int i = tid; i < MT_R * 8; i += 256) {
        const int row = i >> 3, c = (i & 7) << 2;
        cp16_g2l(x + (size_t)Tid[row] * H_DIM + kb + c, Xs + row * XPF + c);
      }
      // Wg/Wu tiles: 32x128 f32 each = 1024 16B-chunks each
      for (int i = tid; i < 32 * 32; i += 256) {
        const int row = i >> 5, c = (i & 31) << 2;
        cp16_g2l(Wg + (size_t)(kb + row) * F_DIM + nb + c, Wgs + row * WPF + c);
        cp16_g2l(Wu + (size_t)(kb + row) * F_DIM + nb + c, Wus + row * WPF + c);
      }
      async_wait0();
      __syncthreads();
      v16bf bg = ldB_lds(Wgs, wid * 16);
      v16bf bu = ldB_lds(Wus, wid * 16);
#pragma unroll
      for (int ms = 0; ms < 4; ++ms) {
        v16bf a = ldA_f32(Xs, XPF, ms * 16);
        accG[ms] = wmma_bf16(a, bg, accG[ms]);
        accU[ms] = wmma_bf16(a, bu, accU[ms]);
      }
    }
    const int n16 = nb + wid * 16;
#pragma unroll
    for (int ms = 0; ms < 4; ++ms)
#pragma unroll
      for (int r = 0; r < 8; ++r) {
        const int lm = ms * 16 + ((lane >> 4) << 3) + r;
        float g = accG[ms][r], u = accU[ms][r];
        float a = g * fast_rcp(1.f + __expf(-g)) * u * Twt[lm];
        Act[lm * AP + n16 + (lane & 15)] = (__bf16)a;
      }
  }
  __syncthreads();

  // ---- phase 2: out[token, :] += act @ Wd  (fp32 HW atomics) ----
  for (int hb = 0; hb < H_DIM; hb += 128) {
    v8f acc[4] = {};
    for (int kb = 0; kb < F_DIM; kb += 32) {
      __syncthreads();
      for (int i = tid; i < 32 * 32; i += 256) {   // Wd tile 32x128 f32
        const int row = i >> 5, c = (i & 31) << 2;
        cp16_g2l(Wd + (size_t)(kb + row) * H_DIM + hb + c, Wst + row * WPF + c);
      }
      async_wait0();
      __syncthreads();
      v16bf b = ldB_lds(Wst, wid * 16);
#pragma unroll
      for (int ms = 0; ms < 4; ++ms) {
        v16bf a = ldA_bf(Act, AP, ms * 16, kb);
        acc[ms] = wmma_bf16(a, b, acc[ms]);
      }
    }
    const int n16 = hb + wid * 16;
#pragma unroll
    for (int ms = 0; ms < 4; ++ms)
#pragma unroll
      for (int r = 0; r < 8; ++r) {
        const int lm = ms * 16 + ((lane >> 4) << 3) + r;
        __hip_atomic_fetch_add(out + (size_t)Tid[lm] * H_DIM + n16 + (lane & 15),
                               acc[ms][r], __ATOMIC_RELAXED,
                               __HIP_MEMORY_SCOPE_AGENT);
      }
  }
}

// ---------------------------------------------------------------------------
extern "C" void kernel_launch(void* const* d_in, const int* in_sizes, int n_in,
                              void* d_out, int out_size, void* d_ws, size_t ws_size,
                              hipStream_t stream) {
  const float* x     = (const float*)d_in[0];   // [1, 2048, 2048]
  const float* wgate = (const float*)d_in[1];   // [32, 2048]
  const float* wg    = (const float*)d_in[2];   // [32, 2048, 1024]
  const float* wu    = (const float*)d_in[3];   // [32, 2048, 1024]
  const float* wd    = (const float*)d_in[4];   // [32, 1024, 2048]
  const float* sg    = (const float*)d_in[5];   // [2048, 2048]
  const float* su    = (const float*)d_in[6];   // [2048, 2048]
  const float* sd    = (const float*)d_in[7];   // [2048, 2048]
  float* out = (float*)d_out;

  char* ws = (char*)d_ws;                        // needs ~513 KB of scratch
  int*   cnt   = (int*)ws;
  int*   list  = (int*)(ws + 256);
  float* wlist = (float*)(ws + 256 + (size_t)E_NUM * T_NUM * 4);

  k_init<<<1, 32, 0, stream>>>(cnt);
  k_gate<<<T_NUM / 8, 256, 0, stream>>>(x, wgate, cnt, list, wlist);
  k_shared<<<T_NUM / MT_S, 256, 0, stream>>>(x, sg, su, sd, out);
  k_routed<<<dim3(E_NUM, T_NUM / MT_R), 256, 0, stream>>>(x, wg, wu, wd,
                                                          cnt, list, wlist, out);
}
